// DECORE_7567732376338
// MI455X (gfx1250) — matrix operand order repaired and checked by reading, compiled, not run on hardware
//
#include <hip/hip_runtime.h>

typedef __attribute__((ext_vector_type(16))) _Float16 v16h;
typedef __attribute__((ext_vector_type(8)))  _Float16 v8h;
typedef __attribute__((ext_vector_type(8)))  float    v8f;

#define DD 32          // link_state_dim
#define WPB 8          // waves per block (256 threads)
#define LDS_STRIDE 40  // padded row stride (halfs) for staging buffers

// ---------- fragment helpers (CDNA5 wave32 WMMA 16x16x32 f16) ----------
// A (16x32 f16): lane L<16 holds row M=L with K = {0..7,16..23}; lane L+16
// holds row M=L with K = {8..15,24..31}; 2 halfs per VGPR.
// B (32x16 f16): lane n<16 holds col N=n, K = {0..7,16..23}; lane n+16 holds col N=n, K = {8..15,24..31}.
// C/D (16x16 f32): vgpr i holds row M=i (lanes 0-15) / M=i+8 (lanes 16-31), col N=lane&15.

__device__ inline v8f wmma(v16h a, v16h b, v8f c) {
  return __builtin_amdgcn_wmma_f32_16x16x32_f16(false, a, false, b, (short)0, c,
                                                false, false);
}

// Load one lane's A slice from a contiguous f16 row of 32 elements (two b128s).
__device__ inline v16h load_A_row16(const _Float16* __restrict__ rowptr, int half) {
  const int kb0 = half ? 8 : 0;
  v8h x0 = *(const v8h*)(rowptr + kb0);
  v8h x1 = *(const v8h*)(rowptr + kb0 + 16);
  v16h a;
#pragma unroll
  for (int i = 0; i < 8; ++i) { a[i] = x0[i]; a[i + 8] = x1[i]; }
  return a;
}

// Load one lane's A slice from a contiguous fp32 row of 32 elements (with cvt).
__device__ inline v16h load_A_row32(const float* __restrict__ rowptr, int half) {
  const int kb0 = half ? 8 : 0;
  float4 x0 = *(const float4*)(rowptr + kb0);
  float4 x1 = *(const float4*)(rowptr + kb0 + 4);
  float4 x2 = *(const float4*)(rowptr + kb0 + 16);
  float4 x3 = *(const float4*)(rowptr + kb0 + 20);
  v16h a;
  a[0]=(_Float16)x0.x;  a[1]=(_Float16)x0.y;  a[2]=(_Float16)x0.z;  a[3]=(_Float16)x0.w;
  a[4]=(_Float16)x1.x;  a[5]=(_Float16)x1.y;  a[6]=(_Float16)x1.z;  a[7]=(_Float16)x1.w;
  a[8]=(_Float16)x2.x;  a[9]=(_Float16)x2.y;  a[10]=(_Float16)x2.z; a[11]=(_Float16)x2.w;
  a[12]=(_Float16)x3.x; a[13]=(_Float16)x3.y; a[14]=(_Float16)x3.z; a[15]=(_Float16)x3.w;
  return a;
}

// Load one lane's A slice from an LDS-staged f16 row (row m of 32 cols).
__device__ inline v16h load_A_lds(const _Float16* row, int half) {
  const int kb0 = half ? 8 : 0;
  v16h a;
#pragma unroll
  for (int i = 0; i < 8; ++i) {
    a[i]     = row[kb0 + i];
    a[i + 8] = row[kb0 + 16 + i];
  }
  return a;
}

// Pre-converted fragment-ordered weights: frag*32 lanes * 16 halfs.
__device__ inline v16h load_frag(const _Float16* __restrict__ fragbuf, int frag, int lane) {
  return *(const v16h*)(fragbuf + (((size_t)frag << 5) + lane) * 16);
}

__device__ inline float sigmf(float x) { return 1.f / (1.f + __expf(-x)); }
__device__ inline float tanhf_fast(float x) {
  float t = __expf(-2.f * fabsf(x));
  float r = (1.f - t) / (1.f + t);
  return copysignf(r, x);
}

// fragment ids
enum {
  F_WM1_K0N0 = 0, F_WM1_K0N1, F_WM1_K1N0, F_WM1_K1N1,
  F_WM2_N0, F_WM2_N1,
  F_WZ_N0, F_WZ_N1, F_UZ_N0, F_UZ_N1,
  F_WR_N0, F_WR_N1, F_UR_N0, F_UR_N1,
  F_WH_N0, F_WH_N1, F_UH_N0, F_UH_N1,
  N_FRAGS
};

// ---------- weight prep: fp32 row-major -> f16 fragment-ordered ----------
__global__ void prep_weights_kernel(const float* __restrict__ Wm1, const float* __restrict__ Wm2,
                                    const float* __restrict__ Wz, const float* __restrict__ Uz,
                                    const float* __restrict__ Wr, const float* __restrict__ Ur,
                                    const float* __restrict__ Wh, const float* __restrict__ Uh,
                                    _Float16* __restrict__ fragbuf) {
  int t = blockIdx.x * blockDim.x + threadIdx.x;
  if (t >= N_FRAGS * 32) return;
  int frag = t >> 5, lane = t & 31;
  const float* W = Wm1;
  int kbase = 0, nbase = 0;
  switch (frag) {
    case F_WM1_K0N0: W = Wm1; kbase = 0;  nbase = 0;  break;
    case F_WM1_K0N1: W = Wm1; kbase = 0;  nbase = 16; break;
    case F_WM1_K1N0: W = Wm1; kbase = 32; nbase = 0;  break;
    case F_WM1_K1N1: W = Wm1; kbase = 32; nbase = 16; break;
    case F_WM2_N0:   W = Wm2; nbase = 0;  break;
    case F_WM2_N1:   W = Wm2; nbase = 16; break;
    case F_WZ_N0:    W = Wz;  nbase = 0;  break;
    case F_WZ_N1:    W = Wz;  nbase = 16; break;
    case F_UZ_N0:    W = Uz;  nbase = 0;  break;
    case F_UZ_N1:    W = Uz;  nbase = 16; break;
    case F_WR_N0:    W = Wr;  nbase = 0;  break;
    case F_WR_N1:    W = Wr;  nbase = 16; break;
    case F_UR_N0:    W = Ur;  nbase = 0;  break;
    case F_UR_N1:    W = Ur;  nbase = 16; break;
    case F_WH_N0:    W = Wh;  nbase = 0;  break;
    case F_WH_N1:    W = Wh;  nbase = 16; break;
    case F_UH_N0:    W = Uh;  nbase = 0;  break;
    case F_UH_N1:    W = Uh;  nbase = 16; break;
  }
  int half = lane >> 4, n = lane & 15;
#pragma unroll
  for (int i = 0; i < 16; ++i) {
    int k = kbase + (half ? 8 : 0) + (i < 8 ? i : i + 8);
    fragbuf[(size_t)t * 16 + i] = (_Float16)W[k * DD + nbase + n];
  }
}

__global__ void zero4_kernel(float4* __restrict__ p, int n4) {
  int i = blockIdx.x * blockDim.x + threadIdx.x;
  if (i < n4) p[i] = make_float4(0.f, 0.f, 0.f, 0.f);
}

__global__ void cvt_h16_kernel(const float* __restrict__ src, _Float16* __restrict__ dst,
                               int n) {
  int i = blockIdx.x * blockDim.x + threadIdx.x;
  if (i < n) dst[i] = (_Float16)src[i];
}

// ---------- message MLP + scatter: one wave = 16 pairs, 6 WMMAs ----------
__global__ __launch_bounds__(WPB * 32)
void msg_kernel(const _Float16* __restrict__ h16, const int* __restrict__ first,
                const int* __restrict__ second, const _Float16* __restrict__ fragbuf,
                const float* __restrict__ bm1, const float* __restrict__ mm1,
                const float* __restrict__ bm2, const float* __restrict__ mm2,
                float* __restrict__ agg, int n_pairs) {
  __shared__ _Float16 stage[WPB][16][LDS_STRIDE];
  const int lane = threadIdx.x & 31;
  const int wave = threadIdx.x >> 5;
  const int tile = blockIdx.x * WPB + wave;
  const int base = tile * 16;
  if (base >= n_pairs) return;  // wave-uniform; EXEC stays all-1s below
  const int half = lane >> 4;
  const int m = lane & 15;

  int p = base + m;
  if (p >= n_pairs) p = n_pairs - 1;
  const int idxf = first[p];
  const int idxs = second[p];

  v16h a_main  = load_A_row16(h16 + (size_t)idxf * DD, half);
  v16h a_neigh = load_A_row16(h16 + (size_t)idxs * DD, half);

  // layer 1: [16x64] @ Wm1[64x32]  (K split: main | neigh)
  v8f c0 = {}, c1 = {};
  c0 = wmma(a_main,  load_frag(fragbuf, F_WM1_K0N0, lane), c0);
  c1 = wmma(a_main,  load_frag(fragbuf, F_WM1_K0N1, lane), c1);
  c0 = wmma(a_neigh, load_frag(fragbuf, F_WM1_K1N0, lane), c0);
  c1 = wmma(a_neigh, load_frag(fragbuf, F_WM1_K1N1, lane), c1);

  // bias + relu + mask, stage as f16 for layer-2 A operand
  const float b0 = bm1[m],      k0 = mm1[m];
  const float b1 = bm1[m + 16], k1 = mm1[m + 16];
#pragma unroll
  for (int i = 0; i < 8; ++i) {
    int row = i + 8 * half;
    float v0 = fmaxf(c0[i] + b0, 0.f) * k0;
    float v1 = fmaxf(c1[i] + b1, 0.f) * k1;
    stage[wave][row][m]      = (_Float16)v0;
    stage[wave][row][m + 16] = (_Float16)v1;
  }
  v16h a_h1 = load_A_lds(&stage[wave][m][0], half);

  // layer 2: [16x32] @ Wm2[32x32]
  v8f d0 = {}, d1 = {};
  d0 = wmma(a_h1, load_frag(fragbuf, F_WM2_N0, lane), d0);
  d1 = wmma(a_h1, load_frag(fragbuf, F_WM2_N1, lane), d1);

  const float b20 = bm2[m],      k20 = mm2[m];
  const float b21 = bm2[m + 16], k21 = mm2[m + 16];
#pragma unroll
  for (int i = 0; i < 8; ++i) {
    int row = i + 8 * half;
    int pr = base + row;
    int dest = __shfl(idxs, row, 32);  // lane `row` holds second[base+row]
    if (pr < n_pairs) {
      float v0 = fmaxf(d0[i] + b20, 0.f) * k20;
      float v1 = fmaxf(d1[i] + b21, 0.f) * k21;
      atomicAdd(&agg[(size_t)dest * DD + m], v0);
      atomicAdd(&agg[(size_t)dest * DD + m + 16], v1);
    }
  }
}

// ---------- GRU update: one wave = 16 links, 12 WMMAs ----------
__global__ __launch_bounds__(WPB * 32)
void gru_kernel(float* __restrict__ h, _Float16* __restrict__ h16,
                const float* __restrict__ agg, const _Float16* __restrict__ fragbuf,
                const float* __restrict__ bz, const float* __restrict__ br,
                const float* __restrict__ bh, int n_links) {
  __shared__ _Float16 stage[WPB][16][LDS_STRIDE];
  const int lane = threadIdx.x & 31;
  const int wave = threadIdx.x >> 5;
  const int tile = blockIdx.x * WPB + wave;
  const int base = tile * 16;
  if (base >= n_links) return;  // wave-uniform
  const int half = lane >> 4;
  const int m = lane & 15;

  int row = base + m;
  if (row >= n_links) row = n_links - 1;

  v16h a_agg = load_A_row32(agg + (size_t)row * DD, half);
  v16h a_h   = load_A_row16(h16 + (size_t)row * DD, half);

  v8f z0 = {}, z1 = {}, r0 = {}, r1 = {};
  z0 = wmma(a_agg, load_frag(fragbuf, F_WZ_N0, lane), z0);
  z1 = wmma(a_agg, load_frag(fragbuf, F_WZ_N1, lane), z1);
  r0 = wmma(a_agg, load_frag(fragbuf, F_WR_N0, lane), r0);
  r1 = wmma(a_agg, load_frag(fragbuf, F_WR_N1, lane), r1);
  z0 = wmma(a_h,   load_frag(fragbuf, F_UZ_N0, lane), z0);
  z1 = wmma(a_h,   load_frag(fragbuf, F_UZ_N1, lane), z1);
  r0 = wmma(a_h,   load_frag(fragbuf, F_UR_N0, lane), r0);
  r1 = wmma(a_h,   load_frag(fragbuf, F_UR_N1, lane), r1);

  const float bz0 = bz[m], bz1 = bz[m + 16];
  const float br0 = br[m], br1 = br[m + 16];

  // r * h -> staged f16 A operand
#pragma unroll
  for (int i = 0; i < 8; ++i) {
    int rr = i + 8 * half;
    int lr = base + rr; if (lr >= n_links) lr = n_links - 1;
    float hv0 = h[(size_t)lr * DD + m];
    float hv1 = h[(size_t)lr * DD + m + 16];
    float rv0 = sigmf(r0[i] + br0);
    float rv1 = sigmf(r1[i] + br1);
    stage[wave][rr][m]      = (_Float16)(rv0 * hv0);
    stage[wave][rr][m + 16] = (_Float16)(rv1 * hv1);
  }
  v16h a_rh = load_A_lds(&stage[wave][m][0], half);

  v8f hh0 = {}, hh1 = {};
  hh0 = wmma(a_agg, load_frag(fragbuf, F_WH_N0, lane), hh0);
  hh1 = wmma(a_agg, load_frag(fragbuf, F_WH_N1, lane), hh1);
  hh0 = wmma(a_rh,  load_frag(fragbuf, F_UH_N0, lane), hh0);
  hh1 = wmma(a_rh,  load_frag(fragbuf, F_UH_N1, lane), hh1);

  const float bh0 = bh[m], bh1 = bh[m + 16];
#pragma unroll
  for (int i = 0; i < 8; ++i) {
    int rr = i + 8 * half;
    int lr = base + rr;
    if (lr < n_links) {
      float hv0 = h[(size_t)lr * DD + m];
      float hv1 = h[(size_t)lr * DD + m + 16];
      float zv0 = sigmf(z0[i] + bz0);
      float zv1 = sigmf(z1[i] + bz1);
      float nh0 = tanhf_fast(hh0[i] + bh0);
      float nh1 = tanhf_fast(hh1[i] + bh1);
      float nv0 = zv0 * hv0 + (1.f - zv0) * nh0;
      float nv1 = zv1 * hv1 + (1.f - zv1) * nh1;
      h[(size_t)lr * DD + m]        = nv0;
      h[(size_t)lr * DD + m + 16]   = nv1;
      h16[(size_t)lr * DD + m]      = (_Float16)nv0;
      h16[(size_t)lr * DD + m + 16] = (_Float16)nv1;
    }
  }
}

// ---------- per-graph segment sum ----------
__global__ void graph_sum_kernel(const float* __restrict__ h, const int* __restrict__ gids,
                                 float* __restrict__ emb, int n_links) {
  int t = blockIdx.x * blockDim.x + threadIdx.x;
  int link = t >> 5;
  int col = t & 31;
  if (link < n_links)
    atomicAdd(&emb[(size_t)gids[link] * DD + col], h[(size_t)link * DD + col]);
}

// ---------- readout MLP (G=64, tiny) ----------
__global__ void readout_kernel(const float* __restrict__ emb,
                               const float* __restrict__ W1, const float* __restrict__ b1,
                               const float* __restrict__ m1,
                               const float* __restrict__ W2, const float* __restrict__ b2,
                               const float* __restrict__ m2,
                               const float* __restrict__ W3, const float* __restrict__ b3,
                               float* __restrict__ out, int G) {
  int g = blockIdx.x * blockDim.x + threadIdx.x;
  if (g >= G) return;
  float e[32];
#pragma unroll
  for (int k = 0; k < 32; ++k) e[k] = emb[g * 32 + k];
  float r1[64];
  for (int n = 0; n < 64; ++n) {
    float s = b1[n];
#pragma unroll
    for (int k = 0; k < 32; ++k) s += e[k] * W1[k * 64 + n];
    r1[n] = fmaxf(s, 0.f) * m1[n];
  }
  float r2[64];
  for (int n = 0; n < 64; ++n) {
    float s = b2[n];
#pragma unroll
    for (int k = 0; k < 64; ++k) s += r1[k] * W2[k * 64 + n];
    r2[n] = fmaxf(s, 0.f) * m2[n];
  }
  float s = b3[0];
#pragma unroll
  for (int k = 0; k < 64; ++k) s += r2[k] * W3[k];
  out[g] = s;
}

extern "C" void kernel_launch(void* const* d_in, const int* in_sizes, int n_in,
                              void* d_out, int out_size, void* d_ws, size_t ws_size,
                              hipStream_t stream) {
  const float* link_state = (const float*)d_in[0];
  const int* first  = (const int*)d_in[1];
  const int* second = (const int*)d_in[2];
  const int* gids   = (const int*)d_in[3];
  const float* Wm1 = (const float*)d_in[5];
  const float* bm1 = (const float*)d_in[6];
  const float* mm1 = (const float*)d_in[7];
  const float* Wm2 = (const float*)d_in[8];
  const float* bm2 = (const float*)d_in[9];
  const float* mm2 = (const float*)d_in[10];
  const float* Wz = (const float*)d_in[11];
  const float* Uz = (const float*)d_in[12];
  const float* bz = (const float*)d_in[13];
  const float* Wr = (const float*)d_in[14];
  const float* Ur = (const float*)d_in[15];
  const float* br = (const float*)d_in[16];
  const float* Wh = (const float*)d_in[17];
  const float* Uh = (const float*)d_in[18];
  const float* bh = (const float*)d_in[19];
  const float* W1 = (const float*)d_in[20];
  const float* b1 = (const float*)d_in[21];
  const float* m1 = (const float*)d_in[22];
  const float* W2 = (const float*)d_in[23];
  const float* b2 = (const float*)d_in[24];
  const float* m2 = (const float*)d_in[25];
  const float* W3 = (const float*)d_in[26];
  const float* b3 = (const float*)d_in[27];

  const int n_links = in_sizes[0] / DD;   // 50000
  const int n_pairs = in_sizes[1];        // 800000
  const int G = in_sizes[21];             // 64 (br1 length)
  const int T = 4;
  const int st = n_links * DD;            // state elements

  // workspace layout (all offsets >=32B aligned)
  float* h_buf = (float*)d_ws;                        // st f32
  float* agg   = h_buf + st;                          // st f32
  float* emb   = agg + st;                            // G*DD f32
  _Float16* fragbuf = (_Float16*)(emb + (size_t)G * DD);  // N_FRAGS*512 f16
  _Float16* h16 = fragbuf + (size_t)N_FRAGS * 512;    // st f16

  hipMemcpyAsync(h_buf, link_state, (size_t)st * sizeof(float),
                 hipMemcpyDeviceToDevice, stream);

  prep_weights_kernel<<<(N_FRAGS * 32 + 255) / 256, 256, 0, stream>>>(
      Wm1, Wm2, Wz, Uz, Wr, Ur, Wh, Uh, fragbuf);
  cvt_h16_kernel<<<(st + 255) / 256, 256, 0, stream>>>(link_state, h16, st);

  const int n_tiles_p = (n_pairs + 15) / 16;
  const int n_tiles_l = (n_links + 15) / 16;
  const int blocks_msg = (n_tiles_p + WPB - 1) / WPB;
  const int blocks_gru = (n_tiles_l + WPB - 1) / WPB;

  for (int t = 0; t < T; ++t) {
    zero4_kernel<<<(st / 4 + 255) / 256, 256, 0, stream>>>((float4*)agg, st / 4);
    msg_kernel<<<blocks_msg, WPB * 32, 0, stream>>>(h16, first, second, fragbuf,
                                                    bm1, mm1, bm2, mm2, agg, n_pairs);
    gru_kernel<<<blocks_gru, WPB * 32, 0, stream>>>(h_buf, h16, agg, fragbuf,
                                                    bz, br, bh, n_links);
  }

  zero4_kernel<<<(G * DD / 4 + 255) / 256, 256, 0, stream>>>((float4*)emb, G * DD / 4);
  graph_sum_kernel<<<((size_t)n_links * DD + 255) / 256, 256, 0, stream>>>(h_buf, gids, emb,
                                                                           n_links);
  readout_kernel<<<(G + 63) / 64, 64, 0, stream>>>(emb, W1, b1, m1, W2, b2, m2, W3, b3,
                                                   (float*)d_out, G);
}
